// GATStateNet_15616501088754
// MI455X (gfx1250) — matrix operand order repaired and checked by reading, not compiled, not run on hardware
//
#include <hip/hip_runtime.h>
#include <hip/hip_bf16.h>

typedef __attribute__((ext_vector_type(16))) _Float16 v16h;
typedef __attribute__((ext_vector_type(8)))  float    v8f;

#define HIDC 64
#define NT_N 100000
#define ND_N 50000
#define ETT_E 800000
#define ETD_E 400000

__device__ __forceinline__ float silu_f(float x) { return x / (1.0f + __expf(-x)); }

// ---------------- utility kernels ----------------

__global__ void k_zero(float* __restrict__ p, long n) {
  long i = (long)blockIdx.x * blockDim.x + threadIdx.x;
  long st = (long)gridDim.x * blockDim.x;
  for (; i < n; i += st) p[i] = 0.0f;
}

// out[r, n] = silu(sum_k x[r,k] * W[k,n] + b[n]); small K (12 or 5)
__global__ void k_stem(const float* __restrict__ x, const float* __restrict__ W,
                       const float* __restrict__ b, float* __restrict__ out,
                       long N, int F) {
  long i = (long)blockIdx.x * blockDim.x + threadIdx.x;
  long st = (long)gridDim.x * blockDim.x;
  long total = N * 64;
  for (; i < total; i += st) {
    int n = (int)(i & 63);
    long r = i >> 6;
    float a = b[n];
    const float* xr = x + r * F;
    for (int k = 0; k < F; ++k) a = fmaf(xr[k], W[k * 64 + n], a);
    out[i] = silu_f(a);
  }
}

// Pack f32 weight W[64,64] (row-major, W[k*64+n]) into 8 WMMA B-fragments
// (kt in 0..1, nt in 0..3) in the 16-bit B-matrix VGPR layout:
// lane L, half h -> K = kt*32 + (h<8 ? (L/16)*8 + h : 16 + (L/16)*8 + h-8), N = nt*16 + (L&15)
__global__ void k_pack_w(const float* __restrict__ W, _Float16* __restrict__ wp) {
  int t = threadIdx.x;            // 256 threads, 1 block
  int f = t >> 5, lane = t & 31;  // f = kt*4 + nt
  int kt = f >> 2, nt = f & 3;
  int g = lane >> 4, col = lane & 15;
  int N = nt * 16 + col;
  _Float16* dst = wp + ((size_t)f * 32 + lane) * 16;
#pragma unroll
  for (int h = 0; h < 16; ++h) {
    int K = kt * 32 + (h < 8 ? g * 8 + h : 16 + g * 8 + (h - 8));
    dst[h] = (_Float16)W[K * 64 + N];
  }
}

// ---------------- WMMA GEMM: out[M,64] = act(scale(A)[M,64] @ W[64,64] + bias) ----------------

#define WMMA_F16(Af, Bf, Cf) \
  __builtin_amdgcn_wmma_f32_16x16x32_f16(false, (Af), false, (Bf), (short)0, (Cf), false, false)

__global__ __launch_bounds__(256)
void k_gemm64(const float* __restrict__ A, const float* __restrict__ cnt,
              const _Float16* __restrict__ wpack, const float* __restrict__ bias,
              float* __restrict__ out, int M, int do_relu, int accum) {
  const int lane = threadIdx.x & 31;
  const int wv   = threadIdx.x >> 5;
  const int base = blockIdx.x * 128 + wv * 16;   // 8 waves/block, 16 rows/wave
  if (base >= M) return;
  const int g   = lane >> 4;
  const int row = base + (lane & 15);            // A layout: lane holds row lane%16
  const bool rowOK = row < M;
  float scale = 1.0f;
  if (cnt != nullptr && rowOK) scale = 1.0f / fmaxf(cnt[row], 1.0f);  // fused mean-div
  const float* arow = A + (size_t)row * 64;

  // A fragments (16x32 f16 layout): halves 0..7 -> K=g*8.., halves 8..15 -> K=16+g*8..
  v16h a0, a1;
#pragma unroll
  for (int h = 0; h < 8; ++h) {
    a0[h]     = (_Float16)(rowOK ? arow[g * 8 + h]      * scale : 0.0f);
    a0[h + 8] = (_Float16)(rowOK ? arow[16 + g * 8 + h] * scale : 0.0f);
    a1[h]     = (_Float16)(rowOK ? arow[32 + g * 8 + h] * scale : 0.0f);
    a1[h + 8] = (_Float16)(rowOK ? arow[48 + g * 8 + h] * scale : 0.0f);
  }

  const v16h* wp = (const v16h*)wpack;  // contiguous 32B fragment per lane
  v8f c0 = {}, c1 = {}, c2 = {}, c3 = {};
  c0 = WMMA_F16(a0, wp[0 * 32 + lane], c0);
  c1 = WMMA_F16(a0, wp[1 * 32 + lane], c1);
  c2 = WMMA_F16(a0, wp[2 * 32 + lane], c2);
  c3 = WMMA_F16(a0, wp[3 * 32 + lane], c3);
  c0 = WMMA_F16(a1, wp[4 * 32 + lane], c0);
  c1 = WMMA_F16(a1, wp[5 * 32 + lane], c1);
  c2 = WMMA_F16(a1, wp[6 * 32 + lane], c2);
  c3 = WMMA_F16(a1, wp[7 * 32 + lane], c3);

  // C/D layout: VGPR v, lane L -> M = (L/16)*8 + v, N = L&15
  const int cc = lane & 15;
  const float bb0 = bias[cc], bb1 = bias[16 + cc], bb2 = bias[32 + cc], bb3 = bias[48 + cc];
#pragma unroll
  for (int v = 0; v < 8; ++v) {
    int r = base + g * 8 + v;
    if (r >= M) continue;
    float* orow = out + (size_t)r * 64;
    float x0 = c0[v] + bb0, x1 = c1[v] + bb1, x2 = c2[v] + bb2, x3 = c3[v] + bb3;
    if (do_relu) {
      x0 = fmaxf(x0, 0.f); x1 = fmaxf(x1, 0.f); x2 = fmaxf(x2, 0.f); x3 = fmaxf(x3, 0.f);
    }
    if (accum) {
      orow[cc] += x0; orow[16 + cc] += x1; orow[32 + cc] += x2; orow[48 + cc] += x3;
    } else {
      orow[cc]  = x0; orow[16 + cc]  = x1; orow[32 + cc]  = x2; orow[48 + cc]  = x3;
    }
  }
}

// ---------------- edge gather / scatter-add (mean aggregation numerator + counts) --------------

__global__ void k_scatter(const float* __restrict__ p, const int* __restrict__ src,
                          const int* __restrict__ dst, const int* __restrict__ mask,
                          float* __restrict__ s, float* __restrict__ cnt, long E) {
  long i = (long)blockIdx.x * blockDim.x + threadIdx.x;
  long st = (long)gridDim.x * blockDim.x;
  long total = E * 16;  // 16 float4 chunks per edge row
  for (; i < total; i += st) {
    long e = i >> 4;
    int c = (int)(i & 15);
    float w = 1.0f;
    if (mask) { w = (float)mask[e]; if (w == 0.0f) continue; }
    int sI = src[e], dI = dst[e];
    const float4 m = ((const float4*)(p + (size_t)sI * 64))[c];
    float* sd = s + (size_t)dI * 64 + c * 4;
    atomicAdd(sd + 0, m.x * w);
    atomicAdd(sd + 1, m.y * w);
    atomicAdd(sd + 2, m.z * w);
    atomicAdd(sd + 3, m.w * w);
    if (c == 0) atomicAdd(cnt + dI, w);
  }
}

// ---------------- residual + silu + LayerNorm (one wave32 per node, 2 feats/lane) --------------

__global__ __launch_bounds__(256)
void k_ln(float* __restrict__ x, const float* __restrict__ h,
          const float* __restrict__ gamma, const float* __restrict__ beta,
          const float* __restrict__ beta_s, float hscale, long N) {
  int lane = threadIdx.x & 31;
  long node = (long)blockIdx.x * 8 + (threadIdx.x >> 5);
  if (node >= N) return;
  float* xr = x + node * 64;
  float v0 = xr[lane], v1 = xr[lane + 32];
  if (h) {
    float bsc = *beta_s;
    const float* hr = h + node * 64;
    v0 += bsc * silu_f(hr[lane]      * hscale);
    v1 += bsc * silu_f(hr[lane + 32] * hscale);
  }
  float sum = v0 + v1;
#pragma unroll
  for (int o = 16; o >= 1; o >>= 1) sum += __shfl_xor(sum, o, 32);
  float mean = sum * 0.015625f;
  float d0 = v0 - mean, d1 = v1 - mean;
  float vs = d0 * d0 + d1 * d1;
#pragma unroll
  for (int o = 16; o >= 1; o >>= 1) vs += __shfl_xor(vs, o, 32);
  float rstd = rsqrtf(vs * 0.015625f + 1e-5f);
  xr[lane]      = d0 * rstd * gamma[lane]      + beta[lane];
  xr[lane + 32] = d1 * rstd * gamma[lane + 32] + beta[lane + 32];
}

// ---------------- column sums (feature means for pooling) ----------------

__global__ void k_colsum(const float* __restrict__ x, float* __restrict__ acc, long N) {
  long i = (long)blockIdx.x * blockDim.x + threadIdx.x;
  long st = (long)gridDim.x * blockDim.x;  // multiple of 64 -> column stays fixed
  int col = (int)(i & 63);
  long total = N * 64;
  float s = 0.f;
  for (; i < total; i += st) s += x[i];
  atomicAdd(acc + col, s);
}

// ---------------- head: means -> MLPs -> LN(8) -> out[72] ----------------

__global__ void k_head(const float* __restrict__ accT, const float* __restrict__ accD,
                       const float* W1t, const float* b1t, const float* W2t, const float* b2t,
                       const float* W1d, const float* b1d, const float* W2d, const float* b2d,
                       const float* g8, const float* bb8, float* __restrict__ out) {
  __shared__ float mt[64], md[64], h1[64], pt[8], pd[8];
  int t = threadIdx.x;  // 64 threads
  mt[t] = accT[t] * (1.0f / (float)NT_N);
  md[t] = accD[t] * (1.0f / (float)ND_N);
  __syncthreads();
  out[t] = mt[t];  // embed[0:64] = xt.mean(0)
  float a = b1t[t];
  for (int k = 0; k < 64; ++k) a = fmaf(mt[k], W1t[k * 64 + t], a);
  h1[t] = silu_f(a);
  __syncthreads();
  if (t < 8) {
    float s2 = b2t[t];
    for (int j = 0; j < 64; ++j) s2 = fmaf(h1[j], W2t[j * 8 + t], s2);
    pt[t] = s2;
  }
  __syncthreads();
  float a2 = b1d[t];
  for (int k = 0; k < 64; ++k) a2 = fmaf(md[k], W1d[k * 64 + t], a2);
  h1[t] = silu_f(a2);
  __syncthreads();
  if (t < 8) {
    float s2 = b2d[t];
    for (int j = 0; j < 64; ++j) s2 = fmaf(h1[j], W2d[j * 8 + t], s2);
    pd[t] = s2;
  }
  __syncthreads();
  if (t == 0) {
    float v[8], mean = 0.f;
    for (int o = 0; o < 8; ++o) { v[o] = pt[o] + pd[o]; mean += v[o]; }
    mean *= 0.125f;
    float var = 0.f;
    for (int o = 0; o < 8; ++o) { float d = v[o] - mean; var += d * d; }
    var *= 0.125f;
    float rstd = rsqrtf(var + 1e-5f);
    for (int o = 0; o < 8; ++o) out[64 + o] = (v[o] - mean) * rstd * g8[o] + bb8[o];
  }
}

// ---------------- host orchestration ----------------

extern "C" void kernel_launch(void* const* d_in, const int* in_sizes, int n_in,
                              void* d_out, int out_size, void* d_ws, size_t ws_size,
                              hipStream_t stream) {
  (void)in_sizes; (void)n_in; (void)out_size; (void)ws_size;
  int ii = 0;
  const float* x_tasks   = (const float*)d_in[ii++];
  const float* x_data    = (const float*)d_in[ii++];
  const int*   e_tt_to   = (const int*)d_in[ii++];
  const int*   e_tt_from = (const int*)d_in[ii++];
  const int*   e_td      = (const int*)d_in[ii++];
  const int*   e_mask    = (const int*)d_in[ii++];
  const float* stem_t_W  = (const float*)d_in[ii++];
  const float* stem_t_b  = (const float*)d_in[ii++];
  const float* stem_d_W  = (const float*)d_in[ii++];
  const float* stem_d_b  = (const float*)d_in[ii++];

  struct SageP { const float *Wp, *bp, *Wl, *bl; };
  struct LayerP {
    SageP tt_to, tt_from, dt, td;
    const float *g_t, *b_t, *g_d, *b_d, *beta_t, *beta_d;
  };
  LayerP L[2];
  for (int l = 0; l < 2; ++l) {
    SageP* sg[4] = {&L[l].tt_to, &L[l].tt_from, &L[l].dt, &L[l].td};
    for (int s = 0; s < 4; ++s) {
      sg[s]->Wp = (const float*)d_in[ii++];
      sg[s]->bp = (const float*)d_in[ii++];
      sg[s]->Wl = (const float*)d_in[ii++];
      sg[s]->bl = (const float*)d_in[ii++];
    }
    L[l].g_t    = (const float*)d_in[ii++];
    L[l].b_t    = (const float*)d_in[ii++];
    L[l].g_d    = (const float*)d_in[ii++];
    L[l].b_d    = (const float*)d_in[ii++];
    L[l].beta_t = (const float*)d_in[ii++];
    L[l].beta_d = (const float*)d_in[ii++];
  }
  const float* fln_t_g = (const float*)d_in[ii++];
  const float* fln_t_b = (const float*)d_in[ii++];
  const float* fln_d_g = (const float*)d_in[ii++];
  const float* fln_d_b = (const float*)d_in[ii++];
  const float* pt_W1 = (const float*)d_in[ii++];
  const float* pt_b1 = (const float*)d_in[ii++];
  const float* pt_W2 = (const float*)d_in[ii++];
  const float* pt_b2 = (const float*)d_in[ii++];
  const float* pd_W1 = (const float*)d_in[ii++];
  const float* pd_b1 = (const float*)d_in[ii++];
  const float* pd_W2 = (const float*)d_in[ii++];
  const float* pd_b2 = (const float*)d_in[ii++];
  const float* mlp_g = (const float*)d_in[ii++];
  const float* mlp_b = (const float*)d_in[ii++];

  char* ws = (char*)d_ws;
  size_t off = 0;
  auto carve = [&](size_t bytes) -> char* {
    char* p = ws + off;
    off += (bytes + 255) & ~(size_t)255;
    return p;
  };
  float*    xt    = (float*)carve((size_t)NT_N * 64 * 4);
  float*    xd    = (float*)carve((size_t)ND_N * 64 * 4);
  float*    pbuf  = (float*)carve((size_t)NT_N * 64 * 4);  // projected features
  float*    sbuf  = (float*)carve((size_t)NT_N * 64 * 4);  // scatter accumulator
  float*    cntb  = (float*)carve((size_t)NT_N * 4);       // degree/weight sums
  float*    h_t   = (float*)carve((size_t)NT_N * 64 * 4);
  float*    h_d   = (float*)carve((size_t)ND_N * 64 * 4);
  _Float16* wpack = (_Float16*)carve(4096 * 2);            // packed WMMA weights
  float*    accT  = (float*)carve(64 * 4);
  float*    accD  = (float*)carve(64 * 4);

  auto gemm = [&](const float* A, const float* cnt, const float* W, const float* bias,
                  float* out, int M, int relu, int accum) {
    k_pack_w<<<1, 256, 0, stream>>>(W, wpack);
    k_gemm64<<<(M + 127) / 128, 256, 0, stream>>>(A, cnt, wpack, bias, out, M, relu, accum);
  };
  auto sage = [&](const float* xsrc, int n_src, const SageP& p,
                  const int* src, const int* dst, long E, const int* mask,
                  int n_dst, float* out, int accum) {
    gemm(xsrc, nullptr, p.Wp, p.bp, pbuf, n_src, 1, 0);     // p = relu(lin_p(x))
    k_zero<<<2048, 256, 0, stream>>>(sbuf, (long)n_dst * 64);
    k_zero<<<256, 256, 0, stream>>>(cntb, (long)n_dst);
    k_scatter<<<2048, 256, 0, stream>>>(pbuf, src, dst, mask, sbuf, cntb, E);
    gemm(sbuf, cntb, p.Wl, p.bl, out, n_dst, 0, accum);     // lin_l(mean aggr)
  };

  // stems
  k_stem<<<2048, 256, 0, stream>>>(x_tasks, stem_t_W, stem_t_b, xt, (long)NT_N, 12);
  k_stem<<<2048, 256, 0, stream>>>(x_data,  stem_d_W, stem_d_b, xd, (long)ND_N, 5);

  const int* s_td = e_td;
  const int* d_td = e_td + ETD_E;
  for (int l = 0; l < 2; ++l) {
    sage(xt, NT_N, L[l].tt_to,   e_tt_to,         e_tt_to + ETT_E,   ETT_E, nullptr, NT_N, h_t, 0);
    sage(xt, NT_N, L[l].tt_from, e_tt_from,       e_tt_from + ETT_E, ETT_E, nullptr, NT_N, h_t, 1);
    sage(xd, ND_N, L[l].dt,      d_td,            s_td,              ETD_E, e_mask,  NT_N, h_t, 1);
    sage(xt, NT_N, L[l].td,      s_td,            d_td,              ETD_E, e_mask,  ND_N, h_d, 0);
    k_ln<<<(NT_N + 7) / 8, 256, 0, stream>>>(xt, h_t, L[l].g_t, L[l].b_t, L[l].beta_t,
                                             1.0f / 3.0f, (long)NT_N);
    k_ln<<<(ND_N + 7) / 8, 256, 0, stream>>>(xd, h_d, L[l].g_d, L[l].b_d, L[l].beta_d,
                                             1.0f, (long)ND_N);
  }

  // final LNs
  k_ln<<<(NT_N + 7) / 8, 256, 0, stream>>>(xt, nullptr, fln_t_g, fln_t_b, nullptr, 1.f, (long)NT_N);
  k_ln<<<(ND_N + 7) / 8, 256, 0, stream>>>(xd, nullptr, fln_d_g, fln_d_b, nullptr, 1.f, (long)ND_N);

  // pooling + head
  k_zero<<<1, 128, 0, stream>>>(accT, 64);
  k_zero<<<1, 128, 0, stream>>>(accD, 64);
  k_colsum<<<1024, 256, 0, stream>>>(xt, accT, (long)NT_N);
  k_colsum<<<1024, 256, 0, stream>>>(xd, accD, (long)ND_N);
  k_head<<<1, 64, 0, stream>>>(accT, accD, pt_W1, pt_b1, pt_W2, pt_b2,
                               pd_W1, pd_b1, pd_W2, pd_b2, mlp_g, mlp_b, (float*)d_out);
}
